// MultiHeadAttention_44478681318131
// MI455X (gfx1250) — matrix operand order repaired and checked by reading, not compiled, not run on hardware
//
#include <hip/hip_runtime.h>

// ---------------------------------------------------------------------------
// CDNA5 (gfx1250) multi-head causal attention.
// bf16 WMMA for all matmuls; GEMM operand tiles staged into LDS by the
// Tensor Data Mover (tensor_load_to_lds, double-buffered, s_wait_tensorcnt).
// Pipeline: cvt->bf16 (weights pre-transposed), QKV TDM+WMMA GEMM (scatter to
// Q/K/Vt), flash attention (online softmax, 16 queries/wave), out-proj GEMM.
// ---------------------------------------------------------------------------

typedef __bf16 bf16;
typedef __attribute__((ext_vector_type(8)))  __bf16 bf16x8;
typedef __attribute__((ext_vector_type(16))) __bf16 bf16x16;
typedef __attribute__((ext_vector_type(8)))  float  v8f;
typedef __attribute__((ext_vector_type(4)))  unsigned int u32x4;
typedef __attribute__((ext_vector_type(8)))  int i32x8;
typedef __attribute__((ext_vector_type(4)))  int i32x4;

#define DMODEL 1024
#define NHEADS 16
#define DKH    64
#define BSZ    4
#define SEQL   2048
#define MROWS  (BSZ * SEQL)        // 8192

static __device__ __forceinline__ bf16 f2bf(float f) {
  unsigned u = __builtin_bit_cast(unsigned, f);
  u += 0x7FFFu + ((u >> 16) & 1u);               // round-to-nearest-even
  unsigned short s = (unsigned short)(u >> 16);
  return __builtin_bit_cast(bf16, s);
}

static __device__ __forceinline__ bf16x8 ld8(const bf16* p) {
  return *(const bf16x8*)p;
}
static __device__ __forceinline__ bf16x16 cat8(bf16x8 lo, bf16x8 hi) {
  return __builtin_shufflevector(lo, hi, 0,1,2,3,4,5,6,7,8,9,10,11,12,13,14,15);
}
static __device__ __forceinline__ v8f wmma_bf(bf16x16 a, bf16x16 b, v8f c) {
  return __builtin_amdgcn_wmma_f32_16x16x32_bf16(
      false, a, false, b, (short)0, c, false, false);
}

static __device__ __forceinline__ unsigned lds_addr32(void* p) {
  return (unsigned)(__SIZE_TYPE__)(__attribute__((address_space(3))) void*)p;
}

// TDM: 2D tile load Global->LDS.  data_size=2B; LDS padding 4 DWORDs (16B)
// after every 16 DWORDs (64B) -> LDS row stride 80B for a 32-elem tile row.
static __device__ __forceinline__ void tdm_load_2d(
    unsigned ldsOff, const bf16* gptr,
    unsigned tensor_d0, unsigned tensor_d1,
    unsigned tile_d0, unsigned tile_d1, unsigned stride0) {
  unsigned long long ga = (unsigned long long)(__SIZE_TYPE__)gptr;
  u32x4 g0;
  g0[0] = 1u;                                          // count=1, user mode
  g0[1] = ldsOff;                                      // LDS byte address
  g0[2] = (unsigned)ga;                                // global_addr[31:0]
  g0[3] = (unsigned)((ga >> 32) & 0x01FFFFFFull) | (2u << 30);  // addr[56:32]|type=2
  i32x8 g1;
  g1[0] = (int)((1u << 16) |                           // data_size = 2B
                (1u << 20) |                           // pad_enable
                (3u << 22) |                           // pad_interval: 16 DW (64B)
                (3u << 25));                           // pad_amount:   4 DW (16B)
  g1[1] = (int)((tensor_d0 & 0xFFFFu) << 16);          // atomic_addr=0 | td0.lo
  g1[2] = (int)((tensor_d0 >> 16) | ((tensor_d1 & 0xFFFFu) << 16));
  g1[3] = (int)((tensor_d1 >> 16) | ((tile_d0 & 0xFFFFu) << 16));
  g1[4] = (int)(tile_d1 & 0xFFFFu);                    // tile_dim1 | tile_dim2=0
  g1[5] = (int)stride0;                                // tensor_dim0_stride lo
  g1[6] = 0;                                           // stride0 hi | stride1 lo
  g1[7] = 0;
  i32x4 z4 = {};
#if defined(__clang_major__) && __clang_major__ >= 23
  i32x8 z8 = {};
  __builtin_amdgcn_tensor_load_to_lds(g0, g1, z4, z4, z8, 0);
#else
  __builtin_amdgcn_tensor_load_to_lds(g0, g1, z4, z4, 0);
#endif
}

// ---------------------------------------------------------------------------
// fp32 -> bf16 conversion (plain and transposing variants)
// ---------------------------------------------------------------------------
__global__ __launch_bounds__(256) void cvt_f32_bf16(const float* __restrict__ s,
                                                    bf16* __restrict__ d, int n) {
  int i = blockIdx.x * 256 + threadIdx.x;
  if (i < n) d[i] = f2bf(s[i]);
}

__global__ __launch_bounds__(256) void cvt_transpose_bf16(
    const float* __restrict__ src, bf16* __restrict__ dst, int K, int N) {
  // src [K][N] f32 -> dst [N][K] bf16
  int i = blockIdx.x * 256 + threadIdx.x;
  if (i < K * N) {
    int k = i / N, n = i % N;
    dst[(size_t)n * K + k] = f2bf(src[i]);
  }
}

// ---------------------------------------------------------------------------
// WMMA GEMM: C[M,N] = A[M,1024] * Bt[N,1024]^T + bias
// block tile 128x128, 8 waves (4x2), wave tile 32x64 (2x4 C tiles), K-step 32.
// A and Bt tiles DMA'd into double-buffered LDS by the TDM; TDM for step i+1
// overlaps the WMMAs of step i.
// mode 0: N=3072, scatter -> Qb[bh,s,dk]*0.125, Kb[bh,s,dk], Vt[bh,dk,s]
// mode 1: N=1024, f32 row-major output
// ---------------------------------------------------------------------------
__global__ __launch_bounds__(256) void gemm_wmma_tdm(
    const bf16* __restrict__ A, const bf16* __restrict__ Bt,
    const float* __restrict__ bias, int N, int mode,
    bf16* __restrict__ Qb, bf16* __restrict__ Kb, bf16* __restrict__ Vt,
    float* __restrict__ Out) {
  __shared__ bf16 As[2][128][40];   // [buf][m][k], row stride 80B (TDM-padded)
  __shared__ bf16 Bs[2][128][40];   // [buf][n][k]

  const int t    = threadIdx.x;
  const int lane = t & 31;
  const int w    = t >> 5;
  const int nb   = N >> 7;                    // 128-wide N blocks
  const int m0   = (blockIdx.x / nb) * 128;
  const int n0   = (blockIdx.x % nb) * 128;
  const int mw   = (w >> 1) * 32;             // wave rows in block tile
  const int nw   = (w & 1) * 64;              // wave cols in block tile
  const int half = lane >> 4;
  const int lr   = lane & 15;

  unsigned asOff[2] = { lds_addr32(&As[0][0][0]), lds_addr32(&As[1][0][0]) };
  unsigned bsOff[2] = { lds_addr32(&Bs[0][0][0]), lds_addr32(&Bs[1][0][0]) };

  const bf16* Ap = A  + (size_t)m0 * 1024;
  const bf16* Bp = Bt + (size_t)n0 * 1024;

  v8f acc[2][4] = {};

  // prologue: DMA k0=0 tiles into buffer 0
  if (w == 0) {
    tdm_load_2d(asOff[0], Ap, 1024, MROWS, 32, 128, 1024);
    tdm_load_2d(bsOff[0], Bp, 1024, (unsigned)N, 32, 128, 1024);
    __builtin_amdgcn_s_wait_tensorcnt(0);
  }
  __syncthreads();

  int p = 0;
  for (int k0 = 0; k0 < 1024; k0 += 32) {
    // overlap: DMA next k-step into the other buffer while we compute
    if (w == 0 && k0 + 32 < 1024) {
      tdm_load_2d(asOff[p ^ 1], Ap + k0 + 32, 1024, MROWS, 32, 128, 1024);
      tdm_load_2d(bsOff[p ^ 1], Bp + k0 + 32, 1024, (unsigned)N, 32, 128, 1024);
    }

    // A frag: lanes 0-15 K{0..7,16..23}; lanes 16-31 K{8..15,24..31}
    const int ca = half * 8;
    bf16x16 a0 = cat8(ld8(&As[p][mw + lr][ca]),      ld8(&As[p][mw + lr][ca + 16]));
    bf16x16 a1 = cat8(ld8(&As[p][mw + 16 + lr][ca]), ld8(&As[p][mw + 16 + lr][ca + 16]));
    // B frag: lanes 0-15 K{0..7,16..23}; lanes 16-31 K{8..15,24..31} (col = lane%16)
    const int cb = half * 8;
#pragma unroll
    for (int j = 0; j < 4; ++j) {
      bf16x16 bj = cat8(ld8(&Bs[p][nw + j * 16 + lr][cb]),
                        ld8(&Bs[p][nw + j * 16 + lr][cb + 16]));
      acc[0][j] = wmma_bf(a0, bj, acc[0][j]);
      acc[1][j] = wmma_bf(a1, bj, acc[1][j]);
    }

    if (w == 0 && k0 + 32 < 1024) __builtin_amdgcn_s_wait_tensorcnt(0);
    __syncthreads();
    p ^= 1;
  }

  // epilogue: C layout row = r + 8*half, col = lane%16
#pragma unroll
  for (int i = 0; i < 2; ++i)
#pragma unroll
    for (int j = 0; j < 4; ++j)
#pragma unroll
      for (int r = 0; r < 8; ++r) {
        int row = m0 + mw + i * 16 + r + half * 8;
        int col = n0 + nw + j * 16 + lr;
        float v = acc[i][j][r] + bias[col];
        if (mode == 1) {
          Out[(size_t)row * DMODEL + col] = v;
        } else {
          int sel = col >> 10, dmc = col & 1023;
          int h = dmc >> 6, d = dmc & 63;
          int b = row >> 11, s = row & 2047;
          size_t bh = (size_t)(b * NHEADS + h);
          if (sel == 0)      Qb[(bh * SEQL + s) * DKH + d] = f2bf(v * 0.125f);
          else if (sel == 1) Kb[(bh * SEQL + s) * DKH + d] = f2bf(v);
          else               Vt[(bh * DKH + d) * SEQL + s] = f2bf(v);
        }
      }
}

// ---------------------------------------------------------------------------
// Flash attention: block = 8 waves; wave owns 16 queries; streams 32-key
// blocks with online softmax. Q pre-scaled by 1/sqrt(dk). Output -> Zb
// [b, s, h*64+d] bf16.
// ---------------------------------------------------------------------------
__global__ __launch_bounds__(256) void flash_attn(
    const bf16* __restrict__ Qb, const bf16* __restrict__ Kb,
    const bf16* __restrict__ Vt, bf16* __restrict__ Zb) {
  __shared__ bf16 Pl[8][16][40];               // per-wave P transpose staging

  const int lane = threadIdx.x & 31;
  const int w    = threadIdx.x >> 5;
  const int bh   = blockIdx.x >> 4;            // 0..63
  const int qblk = blockIdx.x & 15;
  const int q0   = qblk * 128 + w * 16;
  const int half = lane >> 4;
  const int lr   = lane & 15;

  const bf16* Q = Qb + (size_t)bh * SEQL * DKH;
  const bf16* K = Kb + (size_t)bh * SEQL * DKH;
  const bf16* V = Vt + (size_t)bh * DKH * SEQL;

  const int ca = half * 16;
  const bf16* qp = Q + (size_t)(q0 + lr) * DKH;
  bf16x16 aQ0 = cat8(ld8(qp + ca),      ld8(qp + ca + 8));
  bf16x16 aQ1 = cat8(ld8(qp + 32 + ca), ld8(qp + 32 + ca + 8));

  v8f o[4] = {};
  float mrow[8], lrow[8];
#pragma unroll
  for (int r = 0; r < 8; ++r) { mrow[r] = -__builtin_inff(); lrow[r] = 0.f; }

  for (int key0 = 0; key0 < q0 + 16; key0 += 32) {
    const bf16* kp0 = K + (size_t)(key0 + lr) * DKH + half * 16;
    const bf16* kp1 = K + (size_t)(key0 + 16 + lr) * DKH + half * 16;
    bf16x16 bK00 = cat8(ld8(kp0),      ld8(kp0 + 8));
    bf16x16 bK01 = cat8(ld8(kp0 + 32), ld8(kp0 + 40));
    bf16x16 bK10 = cat8(ld8(kp1),      ld8(kp1 + 8));
    bf16x16 bK11 = cat8(ld8(kp1 + 32), ld8(kp1 + 40));

    v8f s0 = {}, s1 = {};
    s0 = wmma_bf(aQ0, bK00, s0); s0 = wmma_bf(aQ1, bK01, s0);
    s1 = wmma_bf(aQ0, bK10, s1); s1 = wmma_bf(aQ1, bK11, s1);

    if (key0 + 31 > q0) {                      // causal mask on diagonal blocks
#pragma unroll
      for (int r = 0; r < 8; ++r) {
        int qr = q0 + r + half * 8;
        if (key0 + lr > qr)      s0[r] = -__builtin_inff();
        if (key0 + 16 + lr > qr) s1[r] = -__builtin_inff();
      }
    }

#pragma unroll
    for (int r = 0; r < 8; ++r) {
      float mx = fmaxf(s0[r], s1[r]);
      mx = fmaxf(mx, __shfl_xor(mx, 1));
      mx = fmaxf(mx, __shfl_xor(mx, 2));
      mx = fmaxf(mx, __shfl_xor(mx, 4));
      mx = fmaxf(mx, __shfl_xor(mx, 8));
      float mn    = fmaxf(mrow[r], mx);
      float alpha = __expf(mrow[r] - mn);
      mrow[r] = mn;
      float p0 = __expf(s0[r] - mn);
      float p1 = __expf(s1[r] - mn);
      float rs = p0 + p1;
      rs += __shfl_xor(rs, 1);
      rs += __shfl_xor(rs, 2);
      rs += __shfl_xor(rs, 4);
      rs += __shfl_xor(rs, 8);
      lrow[r] = alpha * lrow[r] + rs;
      o[0][r] *= alpha; o[1][r] *= alpha; o[2][r] *= alpha; o[3][r] *= alpha;
      Pl[w][r + half * 8][lr]      = f2bf(p0);
      Pl[w][r + half * 8][16 + lr] = f2bf(p1);
    }
    asm volatile("s_wait_dscnt 0" ::: "memory");

    bf16x16 aP = cat8(ld8(&Pl[w][lr][ca]), ld8(&Pl[w][lr][ca + 8]));

#pragma unroll
    for (int j = 0; j < 4; ++j) {
      const bf16* vp = V + (size_t)(j * 16 + lr) * SEQL + key0 + half * 16;
      bf16x16 bV = cat8(ld8(vp), ld8(vp + 8));
      o[j] = wmma_bf(aP, bV, o[j]);
    }
  }

  const int b = bh >> 4, hh = bh & 15;
#pragma unroll
  for (int r = 0; r < 8; ++r) {
    float inv = 1.0f / lrow[r];
    int s = q0 + r + half * 8;
    bf16* zp = Zb + ((size_t)b * SEQL + s) * DMODEL + hh * DKH;
#pragma unroll
    for (int j = 0; j < 4; ++j) zp[j * 16 + lr] = f2bf(o[j][r] * inv);
  }
}

// ---------------------------------------------------------------------------
extern "C" void kernel_launch(void* const* d_in, const int* in_sizes, int n_in,
                              void* d_out, int out_size, void* d_ws, size_t ws_size,
                              hipStream_t stream) {
  (void)in_sizes; (void)n_in; (void)out_size; (void)ws_size;
  const float* x     = (const float*)d_in[0];   // [4,2048,1024]
  const float* w_qkv = (const float*)d_in[1];   // [1024,3072]
  const float* b_qkv = (const float*)d_in[2];   // [3072]
  const float* w_out = (const float*)d_in[3];   // [1024,1024]
  const float* b_out = (const float*)d_in[4];   // [1024]
  float* out = (float*)d_out;

  char* ws = (char*)d_ws;
  bf16* xb     = (bf16*)ws;  ws += (size_t)MROWS * DMODEL * 2;        // 16 MB
  bf16* wqkvT  = (bf16*)ws;  ws += (size_t)3 * DMODEL * DMODEL * 2;   //  6 MB  [3072][1024]
  bf16* woutT  = (bf16*)ws;  ws += (size_t)DMODEL * DMODEL * 2;       //  2 MB  [1024][1024]
  bf16* Qb     = (bf16*)ws;  ws += (size_t)BSZ * NHEADS * SEQL * DKH * 2;
  bf16* Kb     = (bf16*)ws;  ws += (size_t)BSZ * NHEADS * SEQL * DKH * 2;
  bf16* Vtb    = (bf16*)ws;  ws += (size_t)BSZ * NHEADS * DKH * SEQL * 2;
  bf16* zb     = (bf16*)ws;  ws += (size_t)MROWS * DMODEL * 2;        // ~88 MB total

  int nx = MROWS * DMODEL, nwq = DMODEL * 3 * DMODEL, nwo = DMODEL * DMODEL;
  cvt_f32_bf16<<<(nx + 255) / 256, 256, 0, stream>>>(x, xb, nx);
  cvt_transpose_bf16<<<(nwq + 255) / 256, 256, 0, stream>>>(w_qkv, wqkvT, DMODEL, 3 * DMODEL);
  cvt_transpose_bf16<<<(nwo + 255) / 256, 256, 0, stream>>>(w_out, woutT, DMODEL, DMODEL);

  // QKV projection: M=8192, N=3072 (TDM-staged WMMA GEMM)
  gemm_wmma_tdm<<<(MROWS / 128) * (3 * DMODEL / 128), 256, 0, stream>>>(
      xb, wqkvT, b_qkv, 3 * DMODEL, /*mode=*/0, Qb, Kb, Vtb, nullptr);

  // flash attention: 64 (b,h) x 16 query blocks of 128
  flash_attn<<<BSZ * NHEADS * (SEQL / 128), 256, 0, stream>>>(Qb, Kb, Vtb, zb);

  // output projection: M=8192, N=1024 -> f32
  gemm_wmma_tdm<<<(MROWS / 128) * (DMODEL / 128), 256, 0, stream>>>(
      zb, woutT, b_out, DMODEL, /*mode=*/1, nullptr, nullptr, nullptr, out);
}
